// FixedKernelDownsample_39702677684562
// MI455X (gfx1250) — matrix operand order repaired and checked by reading, not compiled, not run on hardware
//
#include <hip/hip_runtime.h>

// Problem constants (from setup_inputs)
#define BATCH 4
#define CIN   128
#define COUT  256
#define TLEN  65536
#define T1    (TLEN/2)   // 32768
#define T2    (TLEN/4)   // 16384

// Tiling
#define TOUT_TILE 32          // output time points per block
#define YTILE     144         // staged y columns (4*32 + reflect halo, 16-aligned)
#define WSTR      136         // LDS stride (bf16 elems) for W tile = 272 B (TDM pad: 64dw + 4dw)
#define XSTR      136         // LDS stride (bf16 elems) for x tile
#define YSTR      147         // LDS stride (f32) for y tile; 8*147%64=24 avoids store conflicts

typedef __attribute__((ext_vector_type(16))) __bf16 v16bf;
typedef __attribute__((ext_vector_type(8)))  float  v8f;
typedef __attribute__((ext_vector_type(4)))  unsigned int v4u;
typedef __attribute__((ext_vector_type(8)))  int v8i;
typedef __attribute__((ext_vector_type(4)))  int v4i;

union FragBF { uint2 q[4]; v16bf v; };

static __device__ __forceinline__ unsigned short f2bf(float f) {
  // round-to-nearest-even fp32 -> bf16
  unsigned u = __float_as_uint(f);
  u += 0x7FFFu + ((u >> 16) & 1u);
  return (unsigned short)(u >> 16);
}

// ---------------------------------------------------------------------------
// Kernel 1: weight-norm prep.  W'[co,ci] = bf16( g[co] * v[co,ci] / ||v[co,:]|| )
// ---------------------------------------------------------------------------
__global__ __launch_bounds__(128) void wn_prep_kernel(
    const float* __restrict__ wv, const float* __restrict__ wg,
    unsigned short* __restrict__ wbf) {
  __shared__ float red[128];
  const int co = blockIdx.x;
  const int tid = threadIdx.x;
  const float val = wv[co * CIN + tid];
  red[tid] = val * val;
  __syncthreads();
  #pragma unroll
  for (int s = 64; s > 0; s >>= 1) {
    if (tid < s) red[tid] += red[tid + s];
    __syncthreads();
  }
  const float scale = wg[co] / sqrtf(red[0]);
  wbf[co * CIN + tid] = f2bf(val * scale);
}

// ---------------------------------------------------------------------------
// Kernel 2: fused 1x1-conv GEMM (bf16 WMMA, f32 accum) + bias + double
// stride-2 reflect downsample, all through LDS (y never touches HBM).
// W' tile staged by the Tensor Data Mover (TDM) with D# padding reproducing
// the bank-padded LDS stride.  Block: 256 threads (8 waves).
// Grid: (T2/32, 2 co-halves, BATCH).
// ---------------------------------------------------------------------------
__global__ __launch_bounds__(256) void fused_conv_ds_kernel(
    const float* __restrict__ x, const unsigned short* __restrict__ wbf,
    const float* __restrict__ bias, float* __restrict__ out) {
  extern __shared__ unsigned char smem[];
  unsigned short* WL = (unsigned short*)smem;              // [128][WSTR] bf16
  unsigned short* XL = WL + 128 * WSTR;                    // [YTILE][XSTR] bf16 (t-major, ci inner)
  float*          YL = (float*)(XL + YTILE * XSTR);        // [128][YSTR] f32

  const int tid   = threadIdx.x;
  const int t0    = blockIdx.x * TOUT_TILE;                // first output t of tile
  const int hbase = blockIdx.y * 128;                      // co half base
  const int b     = blockIdx.z;
  const int ybase = t0 * 4 - 4;                            // first staged global y index

  // ---- stage W' half: 128 co x 128 ci bf16 -> LDS rows padded to 272 B
#if defined(__gfx1250__) && __has_builtin(__builtin_amdgcn_tensor_load_to_lds)
  if (tid < 32) {                                          // one wave issues the TDM op
    const unsigned lds_off = (unsigned)(uintptr_t)(void*)WL;         // LDS aperture: addr[31:0] = LDS byte offset
    const unsigned long long ga =
        (unsigned long long)(uintptr_t)(wbf + (size_t)hbase * CIN);  // tile start in global
    v4u g0;
    g0[0] = 1u;                                            // count=1 (valid user D#), no gather
    g0[1] = lds_off;                                       // lds_addr
    g0[2] = (unsigned)(ga & 0xFFFFFFFFu);                  // global_addr[31:0]
    g0[3] = (unsigned)((ga >> 32) & 0x1FFFFFFu)            // global_addr[56:32]
          | (2u << 30);                                    // type=2 ("image")
    v8i g1;
    g1[0] = (1 << 16)                                      // data_size=1 -> 2 bytes
          | (1 << 20)                                      // pad_enable
          | (5 << 22)                                      // pad_interval: every 64 DWORDs (256 B)
          | (3 << 25);                                     // pad_amount: 4 DWORDs (16 B) -> stride 272 B
    g1[1] = (CIN << 16);                                   // tensor_dim0[15:0]=128 (bits 79:48)
    g1[2] = (128 << 16);                                   // tensor_dim0 hi=0 | tensor_dim1[15:0]=128
    g1[3] = (CIN << 16);                                   // tensor_dim1 hi=0 | tile_dim0=128
    g1[4] = 128;                                           // tile_dim1=128, tile_dim2=0 (2D)
    g1[5] = CIN;                                           // tensor_dim0_stride[31:0]=128 elems
    g1[6] = 0;                                             // stride hi / dim1_stride (unused, 2D)
    g1[7] = 0;
    const v4i z4 = {0, 0, 0, 0};                           // groups 2/3 unused (<=2D tensor)
    const v8i z8 = {0, 0, 0, 0, 0, 0, 0, 0};               // extra group (clang-23 6-arg form)
    __builtin_amdgcn_tensor_load_to_lds(g0, g1, z4, z4, z8, 0);
    __builtin_amdgcn_s_wait_tensorcnt(0);                  // TENSORcnt==0 before the barrier
  }
#else
  {
    const unsigned short* wsrc = wbf + (size_t)hbase * CIN;
    for (int idx = tid; idx < 128 * 16; idx += 256) {      // 8 iters
      const int co = idx >> 4, cg = idx & 15;
      uint4 d = *(const uint4*)(wsrc + co * CIN + cg * 8);
      unsigned short* dst = WL + co * WSTR + cg * 8;
      *(uint2*)(dst)     = make_uint2(d.x, d.y);
      *(uint2*)(dst + 4) = make_uint2(d.z, d.w);
    }
  }
#endif

  // ---- stage x tile transposed to [t][ci], converting f32 -> bf16 (VALU)
  {
    const float* xb = x + (size_t)b * CIN * TLEN;
    for (int idx = tid; idx < YTILE * 32; idx += 256) {    // 18 iters
      const int t  = idx % YTILE;                          // lane-consecutive -> coalesced
      const int cq = idx / YTILE;                          // ci quad 0..31
      int tg = ybase + t;                                  // halo may poke outside; clamp
      tg = tg < 0 ? 0 : (tg > TLEN - 1 ? TLEN - 1 : tg);   // (clamped vals never read)
      const float* xp = xb + (size_t)(cq * 4) * TLEN + tg;
      const float v0 = xp[0];
      const float v1 = xp[TLEN];
      const float v2 = xp[2 * TLEN];
      const float v3 = xp[3 * TLEN];
      uint2 p;
      p.x = (unsigned)f2bf(v0) | ((unsigned)f2bf(v1) << 16);
      p.y = (unsigned)f2bf(v2) | ((unsigned)f2bf(v3) << 16);
      *(uint2*)(XL + t * XSTR + cq * 4) = p;
    }
  }
  __syncthreads();

  // ---- GEMM: wave w owns co rows [16w,16w+16); 9 t-tiles x (K=128 in 4x32)
  {
    const int lane = tid & 31;
    const int wid  = tid >> 5;
    const int m    = lane & 15;
    const int h16  = lane >> 4;                            // lane half select

    FragBF A[4];                                           // A 16x32 bf16 layout:
    #pragma unroll                                         // elems0-7:K=h16*8+0..7, elems8-15:K=16+h16*8+0..7
    for (int kk = 0; kk < 4; ++kk) {
      const unsigned short* ap = WL + (wid * 16 + m) * WSTR + kk * 32 + h16 * 8;
      A[kk].q[0] = *(const uint2*)(ap);
      A[kk].q[1] = *(const uint2*)(ap + 4);
      A[kk].q[2] = *(const uint2*)(ap + 16);
      A[kk].q[3] = *(const uint2*)(ap + 20);
    }
    for (int tt = 0; tt < 9; ++tt) {
      v8f c = {0.f, 0.f, 0.f, 0.f, 0.f, 0.f, 0.f, 0.f};
      #pragma unroll
      for (int kk = 0; kk < 4; ++kk) {
        FragBF Bf;                                         // B 32x16: col=lane&15, elems0-7:K=h16*8+0..7, elems8-15:K=16+h16*8+0..7
        const unsigned short* bp = XL + (tt * 16 + m) * XSTR + kk * 32 + h16 * 8;
        Bf.q[0] = *(const uint2*)(bp);
        Bf.q[1] = *(const uint2*)(bp + 4);
        Bf.q[2] = *(const uint2*)(bp + 16);
        Bf.q[3] = *(const uint2*)(bp + 20);
        c = __builtin_amdgcn_wmma_f32_16x16x32_bf16(
            /*neg_a=*/false, A[kk].v, /*neg_b=*/false, Bf.v,
            /*c_mod=*/(short)0, c, /*reuse_a=*/false, /*reuse_b=*/false);
      }
      #pragma unroll
      for (int r = 0; r < 8; ++r)                          // D: VGPR r -> M = h16*8 + r
        YL[(wid * 16 + h16 * 8 + r) * YSTR + tt * 16 + m] = c[r];
    }
  }
  __syncthreads();

  // ---- epilogue: exact two-stage stride-2 downsample with reflect, + bias
  {
    float* yout = out + ((size_t)b * COUT + hbase) * T2;
    for (int idx = tid; idx < 128 * TOUT_TILE; idx += 256) { // 16 iters
      const int co = idx >> 5;
      const int t  = idx & 31;
      const int tg = t0 + t;
      const float* yrow = YL + co * YSTR;
      float acc = 0.f;
      #pragma unroll
      for (int j = 0; j < 4; ++j) {
        int u = 2 * tg - 1 + j;                            // stage-2 tap (reflect in y1 space)
        u = u < 0 ? -u : (u > T1 - 1 ? 2 * (T1 - 1) - u : u);
        float y1 = 0.f;
        #pragma unroll
        for (int k = 0; k < 4; ++k) {
          int mm = 2 * u - 1 + k;                          // stage-1 tap (reflect in y space)
          mm = mm < 0 ? -mm : (mm > TLEN - 1 ? 2 * (TLEN - 1) - mm : mm);
          const float wk = (k == 0 || k == 3) ? 0.125f : 0.375f;
          y1 += wk * yrow[mm - ybase];
        }
        const float wj = (j == 0 || j == 3) ? 0.125f : 0.375f;
        acc += wj * y1;
      }
      yout[(size_t)co * T2 + tg] = acc + bias[hbase + co]; // taps sum to 1 -> bias commutes
    }
  }
}

// ---------------------------------------------------------------------------
extern "C" void kernel_launch(void* const* d_in, const int* in_sizes, int n_in,
                              void* d_out, int out_size, void* d_ws, size_t ws_size,
                              hipStream_t stream) {
  const float* x    = (const float*)d_in[0];   // [4,128,65536]
  const float* wv   = (const float*)d_in[1];   // [256,128,1]
  const float* wg   = (const float*)d_in[2];   // [256,1,1]
  const float* bias = (const float*)d_in[3];   // [256]
  float* out = (float*)d_out;                  // [4,256,16384]
  unsigned short* wbf = (unsigned short*)d_ws; // 256*128 bf16 = 64 KB scratch

  wn_prep_kernel<<<COUT, 128, 0, stream>>>(wv, wg, wbf);

  const size_t smem = (size_t)(128 * WSTR + YTILE * XSTR) * sizeof(unsigned short)
                    + (size_t)128 * YSTR * sizeof(float);  // ~146 KB < 160 KB (2 blocks/WGP)
  dim3 grid(T2 / TOUT_TILE, 2, BATCH);                     // 512 x 2 x 4 = 4096 blocks
  fused_conv_ds_kernel<<<grid, 256, smem, stream>>>(x, wbf, bias, out);
}
